// MoERouter_34385508172481
// MI455X (gfx1250) — matrix operand order repaired and checked by reading, not compiled, not run on hardware
//
#include <hip/hip_runtime.h>
#include <hip/hip_bf16.h>
#include <math.h>

typedef __attribute__((ext_vector_type(16))) __bf16 v16bf;
typedef __attribute__((ext_vector_type(8)))  float  v8f;
typedef __attribute__((ext_vector_type(4)))  float  v4f;

#define N_TOK 16384
#define DDIM  4096
#define NEXP  64
#define TOPK  8
#define KTILES (DDIM / 32)      // 128 K-chunks of 32
#define WAVES_PER_BLOCK 4

// ---------------------------------------------------------------------------
// Kernel 0: zero the bincount segment of d_out (it is accumulated atomically).
// ---------------------------------------------------------------------------
__global__ void zero_counts_kernel(int* counts) {
    if (threadIdx.x < NEXP) counts[threadIdx.x] = 0;
}

// ---------------------------------------------------------------------------
// Kernel 1: stage W (4096x64 fp32, row-major) into bf16 WMMA B-fragment layout
// in d_ws. Fragment (kt, nt) covers K rows kt*32..kt*32+31, experts nt*16..+15.
// CDNA5 16-bit B 32x16 layout: lanes 0-15 hold K={0..7,16..23} for column N=lane,
// lanes 16-31 hold K={8..15,24..31} for column N=lane-16 (same K order as A).
// Per lane: 16 bf16 = 32 contiguous bytes.
// ---------------------------------------------------------------------------
__global__ void stage_W_kernel(const float* __restrict__ W, __bf16* __restrict__ wsB) {
    int idx = blockIdx.x * blockDim.x + threadIdx.x;   // 0 .. DDIM*NEXP-1
    if (idx >= DDIM * NEXP) return;
    int j    = idx & 15;          // element within lane (K offset in half)
    int lane = (idx >> 4) & 31;
    int frag = idx >> 9;          // kt*4 + nt
    int nt   = frag & 3;
    int kt   = frag >> 2;
    int n = nt * 16 + (lane & 15);
    int k = kt * 32 + ((lane >> 4) ? 8 : 0) + (j & 7) + ((j >> 3) ? 16 : 0);
    wsB[idx] = (__bf16)W[k * NEXP + n];
}

// ---------------------------------------------------------------------------
// Kernel 2: main router. One wave per 16-token tile.
// ---------------------------------------------------------------------------
__launch_bounds__(WAVES_PER_BLOCK * 32)
__global__ void router_main_kernel(const float* __restrict__ x,
                                   const float* __restrict__ noise,
                                   const float* __restrict__ score_bias,
                                   const __bf16* __restrict__ wsB,
                                   float* __restrict__ out) {
    const int lane      = threadIdx.x & 31;
    const int waveInBlk = threadIdx.x >> 5;
    const int wave      = blockIdx.x * WAVES_PER_BLOCK + waveInBlk;
    const int m0        = wave * 16;                       // first token of tile

    __shared__ float lds[WAVES_PER_BLOCK][16 * NEXP];      // 4 KB per wave
    float* tile = lds[waveInBlk];

    // ---- GEMM: logits[m0..m0+15][0..63] = jitter(x) @ W ----
    v8f c0 = {}, c1 = {}, c2 = {}, c3 = {};

    // A-fragment addressing (16-bit A 16x32 layout):
    //   lanes 0-15 : row M = lane,    K in {0..7, 16..23} of the chunk
    //   lanes 16-31: row M = lane-16, K in {8..15, 24..31}
    const int row   = m0 + (lane & 15);
    const int khalf = (lane >> 4) ? 8 : 0;
    const float* __restrict__ xrow = x     + (size_t)row * DDIM;
    const float* __restrict__ nrow = noise + (size_t)row * DDIM;

    for (int kt = 0; kt < KTILES; ++kt) {
        const int kb = kt * 32 + khalf;
        // prefetch next chunk of the 512MB HBM stream
        if (kt + 1 < KTILES) {
            __builtin_prefetch(xrow + kb + 32, 0, 1);
            __builtin_prefetch(nrow + kb + 32, 0, 1);
        }
        v4f x0 = *(const v4f*)(xrow + kb);
        v4f x1 = *(const v4f*)(xrow + kb + 4);
        v4f x2 = *(const v4f*)(xrow + kb + 16);
        v4f x3 = *(const v4f*)(xrow + kb + 20);
        v4f n0 = *(const v4f*)(nrow + kb);
        v4f n1 = *(const v4f*)(nrow + kb + 4);
        v4f n2 = *(const v4f*)(nrow + kb + 16);
        v4f n3 = *(const v4f*)(nrow + kb + 20);

        // jitter: x * (0.99 + noise * 0.02)
        v16bf a;
#pragma unroll
        for (int i = 0; i < 4; ++i) {
            a[i]      = (__bf16)(x0[i] * fmaf(n0[i], 0.02f, 0.99f));
            a[i + 4]  = (__bf16)(x1[i] * fmaf(n1[i], 0.02f, 0.99f));
            a[i + 8]  = (__bf16)(x2[i] * fmaf(n2[i], 0.02f, 0.99f));
            a[i + 12] = (__bf16)(x3[i] * fmaf(n3[i], 0.02f, 0.99f));
        }

        // B-fragments: contiguous 32B per lane from staged layout
        const __bf16* bbase = wsB + ((size_t)(kt * 4) * 512) + lane * 16;
        v16bf b0 = *(const v16bf*)(bbase);
        v16bf b1 = *(const v16bf*)(bbase + 512);
        v16bf b2 = *(const v16bf*)(bbase + 1024);
        v16bf b3 = *(const v16bf*)(bbase + 1536);

        c0 = __builtin_amdgcn_wmma_f32_16x16x32_bf16(false, a, false, b0, (short)0, c0, false, false);
        c1 = __builtin_amdgcn_wmma_f32_16x16x32_bf16(false, a, false, b1, (short)0, c1, false, false);
        c2 = __builtin_amdgcn_wmma_f32_16x16x32_bf16(false, a, false, b2, (short)0, c2, false, false);
        c3 = __builtin_amdgcn_wmma_f32_16x16x32_bf16(false, a, false, b3, (short)0, c3, false, false);
    }

    // ---- spill C to LDS. C layout: lanes 0-15 N=lane, VGPR v -> M=v; lanes
    //      16-31 N=lane-16, VGPR v -> M=v+8.
    const int cn   = lane & 15;
    const int rofs = (lane >> 4) ? 8 : 0;
#pragma unroll
    for (int v = 0; v < 8; ++v) {
        tile[(v + rofs) * NEXP +  0 + cn] = c0[v];
        tile[(v + rofs) * NEXP + 16 + cn] = c1[v];
        tile[(v + rofs) * NEXP + 32 + cn] = c2[v];
        tile[(v + rofs) * NEXP + 48 + cn] = c3[v];
    }
    __syncthreads();

    // ---- output segment pointers (tuple concatenated flat, 4-byte elems) ----
    float* logits = out;
    float* scores = out + (size_t)N_TOK * NEXP;
    float* wts    = out + 2 * (size_t)N_TOK * NEXP;
    int*   inds   = (int*)(out + 2 * (size_t)N_TOK * NEXP + (size_t)N_TOK * TOPK);
    int*   counts = (int*)(out + 2 * (size_t)N_TOK * NEXP + 2 * (size_t)N_TOK * TOPK);

    // coalesced logits write (1024 floats per tile)
    for (int i = lane; i < 16 * NEXP; i += 32)
        logits[(size_t)m0 * NEXP + i] = tile[i];
    __syncthreads();

    // ---- per-token epilogue: lanes 0-15 each own one token ----
    if (lane < 16) {
        const int t = m0 + lane;
        float* rowp = tile + lane * NEXP;

        // softmax (max-subtracted)
        float mx = -INFINITY;
#pragma unroll
        for (int e = 0; e < NEXP; ++e) mx = fmaxf(mx, rowp[e]);
        float sum = 0.0f;
#pragma unroll
        for (int e = 0; e < NEXP; ++e) {
            float v = __expf(rowp[e] - mx);
            rowp[e] = v;
            sum += v;
        }
        const float inv = 1.0f / sum;
#pragma unroll
        for (int e = 0; e < NEXP; ++e) rowp[e] *= inv;   // rowp now = scores

        // scores out (16 x float4 per token)
#pragma unroll
        for (int e = 0; e < NEXP; e += 4) {
            v4f s = { rowp[e], rowp[e + 1], rowp[e + 2], rowp[e + 3] };
            *(v4f*)(scores + (size_t)t * NEXP + e) = s;
        }

        // bias-corrected top-8: select on scores + score_bias, weight = score
        unsigned long long used = 0ull;
        float w[TOPK];
        int   id[TOPK];
#pragma unroll
        for (int k = 0; k < TOPK; ++k) {
            float best = -INFINITY;
            int   bi   = 0;
            for (int e = 0; e < NEXP; ++e) {
                if (used & (1ull << e)) continue;
                float v = rowp[e] + score_bias[e];
                if (v > best) { best = v; bi = e; }
            }
            used |= (1ull << bi);
            id[k] = bi;
            w[k]  = rowp[bi];
        }
        float nrm = 0.0f;
#pragma unroll
        for (int k = 0; k < TOPK; ++k) nrm += w[k] * w[k];
        const float rinv = rsqrtf(nrm);
#pragma unroll
        for (int k = 0; k < TOPK; ++k) {
            wts[(size_t)t * TOPK + k]  = w[k] * rinv;
            inds[(size_t)t * TOPK + k] = id[k];
            atomicAdd(&counts[id[k]], 1);
        }
    }
}

// ---------------------------------------------------------------------------
extern "C" void kernel_launch(void* const* d_in, const int* in_sizes, int n_in,
                              void* d_out, int out_size, void* d_ws, size_t ws_size,
                              hipStream_t stream) {
    const float* x          = (const float*)d_in[0];   // (16384, 4096)
    const float* W          = (const float*)d_in[1];   // (4096, 64)
    const float* score_bias = (const float*)d_in[2];   // (64,)
    const float* noise      = (const float*)d_in[3];   // (16384, 4096)
    float* out = (float*)d_out;

    __bf16* wsB = (__bf16*)d_ws;                       // 4096*64*2 = 512 KB

    int* counts = (int*)(out + 2 * (size_t)N_TOK * NEXP + 2 * (size_t)N_TOK * TOPK);
    zero_counts_kernel<<<1, 64, 0, stream>>>(counts);

    stage_W_kernel<<<(DDIM * NEXP + 255) / 256, 256, 0, stream>>>(W, wsB);

    const int nWaves  = N_TOK / 16;                    // 1024
    const int nBlocks = nWaves / WAVES_PER_BLOCK;      // 256
    router_main_kernel<<<nBlocks, WAVES_PER_BLOCK * 32, 0, stream>>>(
        x, noise, score_bias, wsB, out);
}
